// MMSingleStreamBlockReFuser_V2_15324443312836
// MI455X (gfx1250) — matrix operand order repaired and checked by reading, not compiled, not run on hardware
//
#include <hip/hip_runtime.h>
#include <math.h>

// ---------------------------------------------------------------------------
// MI455X (gfx1250) fused single-stream DiT block.
//
// Roofline: W1(264MB)+W2(189MB)+ref_Wkv(72MB)+mod_W(113MB) fp32 weight reads
// dominate (~640MB) vs ~640 GFLOP of math -> bandwidth bound (~27us at
// 23.3TB/s). Strategy: read fp32 weights once (GEMM grid is M-fastest so all
// M-tiles of an N-column group run together and W columns hit 192MB L2),
// convert to bf16 in registers, and run every matmul on
// v_wmma_f32_16x16x32_bf16. GEMM LDS tiles are double-buffered (1 barrier /
// K-step, 2 WMMAs per wave per step); bf16 tiles that need no conversion are
// staged with global_load_async_to_lds_b128 (ASYNCcnt-tracked DMA).
// ---------------------------------------------------------------------------

typedef __bf16 bf16x16 __attribute__((ext_vector_type(16)));
typedef float  f32x8   __attribute__((ext_vector_type(8)));
typedef float  f32x4   __attribute__((ext_vector_type(4)));
typedef unsigned short u16x8 __attribute__((ext_vector_type(8)));

union ABFrag { bf16x16 v; u16x8 h[2]; };

#define HID   3072
#define SEQ   2304
#define IMG   2048
#define NH    24
#define HD    128
#define MLP   12288
#define N1    (3 * HID + MLP)   // 21504
#define REF   768
#define NKV   (2 * HID)         // 6144
#define CAT   (HID + MLP)       // 15360

#if defined(__AMDGCN__) && __has_builtin(__builtin_amdgcn_global_load_async_to_lds_b128)
#define HAVE_ASYNC_LDS 1
#else
#define HAVE_ASYNC_LDS 0
#endif

// 16-byte int vector, matching the builtin's parameter type exactly.
typedef int v4i_vs __attribute__((vector_size(16)));
typedef __attribute__((address_space(1))) v4i_vs* gp4_t;  // global int4*
typedef __attribute__((address_space(3))) v4i_vs* lp4_t;  // LDS int4*

// 16-byte global -> LDS copy; async DMA path on CDNA5 (no VGPR round trip).
__device__ __forceinline__ void copy16_to_lds(const unsigned short* gsrc,
                                              unsigned short* ldst) {
#if HAVE_ASYNC_LDS
  __builtin_amdgcn_global_load_async_to_lds_b128(
      (gp4_t)(v4i_vs*)(unsigned short*)gsrc, (lp4_t)(v4i_vs*)ldst, 0, 0);
#else
  *(u16x8*)ldst = *(const u16x8*)gsrc;
#endif
}

__device__ __forceinline__ void async_copy_wait() {
#if HAVE_ASYNC_LDS
#if __has_builtin(__builtin_amdgcn_s_wait_asynccnt)
  __builtin_amdgcn_s_wait_asynccnt(0);
#else
  asm volatile("s_wait_asynccnt 0" ::: "memory");
#endif
#endif
}

__device__ __forceinline__ unsigned short f2bf(float f) {
  union { float f; unsigned u; } v; v.f = f;
  unsigned r = v.u + 0x7FFFu + ((v.u >> 16) & 1u);
  return (unsigned short)(r >> 16);
}

__device__ __forceinline__ float geluf(float x) {
  return 0.5f * x * (1.0f + tanhf(0.7978845608028654f * (x + 0.044715f * x * x * x)));
}

// ------------------------------ tiny kernels -------------------------------

__global__ void silu_kernel(const float* __restrict__ vec, float* __restrict__ sv) {
  int i = blockIdx.x * 256 + threadIdx.x;
  float x = vec[i];
  sv[i] = x / (1.0f + __expf(-x));
}

__global__ void mod_gemv_kernel(const float* __restrict__ sv,
                                const float* __restrict__ modW,
                                const float* __restrict__ modb,
                                float* __restrict__ mod) {
  int o = blockIdx.x * 256 + threadIdx.x;  // 9216 outputs
  float acc = 0.f;
  for (int k = 0; k < HID; ++k) acc += sv[k] * modW[(size_t)k * (3 * HID) + o];
  mod[o] = acc + modb[o];
}

__global__ __launch_bounds__(256)
void ln_mod_kernel(const float* __restrict__ x, const float* __restrict__ mod,
                   unsigned short* __restrict__ xmod) {
  int s = blockIdx.x, t = threadIdx.x;
  __shared__ float r1[256], r2[256];
  const float* row = x + (size_t)s * HID;
  float sum = 0.f, ss = 0.f;
  for (int i = t; i < HID; i += 256) { float v = row[i]; sum += v; ss += v * v; }
  r1[t] = sum; r2[t] = ss; __syncthreads();
  for (int off = 128; off > 0; off >>= 1) {
    if (t < off) { r1[t] += r1[t + off]; r2[t] += r2[t + off]; }
    __syncthreads();
  }
  float mu = r1[0] * (1.0f / HID);
  float var = r2[0] * (1.0f / HID) - mu * mu;
  float rstd = rsqrtf(var + 1e-6f);
  const float* shift = mod;
  const float* scale = mod + HID;
  for (int i = t; i < HID; i += 256) {
    float v = (row[i] - mu) * rstd * (1.0f + scale[i]) + shift[i];
    xmod[(size_t)s * HID + i] = f2bf(v);
  }
}

__global__ void f32_to_bf16_kernel(const float* __restrict__ in,
                                   unsigned short* __restrict__ out) {
  int i = blockIdx.x * 256 + threadIdx.x;
  out[i] = f2bf(in[i]);
}

// ---------------------- generic bf16-WMMA GEMM -----------------------------
// C[M,N] = A(bf16)[M,K] @ B(fp32->bf16)[K,N] + bias, fused epilogues.
// BM=64 BN=64 BK=32, 8 waves as 4(M) x 2(N); each wave owns a 16x32 tile
// (2 WMMAs / K-step, A fragment reused). LDS double-buffered: one barrier
// per K-step; A tile staged via async-to-LDS, B tile converted in VGPRs.

__global__ __launch_bounds__(256)
void gemm_bf16_wmma_kernel(const unsigned short* __restrict__ A,
                           const float* __restrict__ B,
                           const float* __restrict__ bias,
                           int M, int N, int K, int mode,
                           float* __restrict__ outf,
                           const float* __restrict__ xres,
                           const float* __restrict__ gate,
                           float* __restrict__ h_qkv,
                           unsigned short* __restrict__ A2) {
  __shared__ __align__(16) unsigned short As[2][64][32];   // [buf][m][k]
  __shared__ __align__(16) unsigned short BsT[2][64][32];  // [buf][n][k]

  const int tid = threadIdx.x;
  const int wave = tid >> 5, lane = tid & 31;
  const int half = lane >> 4, l16 = lane & 15;
  const int wm = wave >> 1, wn = wave & 1;
  const int rowBase = blockIdx.x * 64;
  const int colBase = blockIdx.y * 64;

  f32x8 acc0, acc1;
  for (int r = 0; r < 8; ++r) { acc0[r] = 0.f; acc1[r] = 0.f; }

  // A stage: 64x32 ushorts = 2048, 8/thread (one 16B async copy)
  const int aIdx = tid * 8;
  const int ar_ = aIdx >> 5, ac_ = aIdx & 31;
  // B stage: 32x64 fp32 = 2048, 8/thread (two f32x4 loads, 8 bf16 LDS stores)
  const int bIdx = tid * 8;
  const int br_ = bIdx >> 6, bc_ = bIdx & 63;

  const int nk = K / 32;

#define GEMM_STAGE(KT, BUF)                                                    \
  do {                                                                         \
    int k0s = (KT) * 32;                                                       \
    copy16_to_lds(A + (size_t)(rowBase + ar_) * K + k0s + ac_,                 \
                  &As[BUF][ar_][ac_]);                                         \
    f32x4 bv0 = *(const f32x4*)(B + (size_t)(k0s + br_) * N + colBase + bc_);  \
    f32x4 bv1 =                                                                \
        *(const f32x4*)(B + (size_t)(k0s + br_) * N + colBase + bc_ + 4);      \
    for (int i = 0; i < 4; ++i) {                                              \
      BsT[BUF][bc_ + i][br_] = f2bf(bv0[i]);                                   \
      BsT[BUF][bc_ + 4 + i][br_] = f2bf(bv1[i]);                               \
    }                                                                          \
  } while (0)

  GEMM_STAGE(0, 0);

  for (int kt = 0; kt < nk; ++kt) {
    const int buf = kt & 1;
    async_copy_wait();
    __syncthreads();
    if (kt + 1 < nk) {
      GEMM_STAGE(kt + 1, buf ^ 1);
      if (kt + 2 < nk)  // pull the W tile after next toward L2
        __builtin_prefetch(B + (size_t)((kt + 2) * 32 + br_) * N + colBase + bc_,
                           0, 2);
    }
    ABFrag a, b0, b1;
    const unsigned short* arow = &As[buf][wm * 16 + l16][0];
    a.h[0] = *(const u16x8*)(arow + 8 * half);
    a.h[1] = *(const u16x8*)(arow + 16 + 8 * half);
    const unsigned short* brow0 = &BsT[buf][wn * 32 + l16][8 * half];
    b0.h[0] = *(const u16x8*)(brow0);
    b0.h[1] = *(const u16x8*)(brow0 + 16);
    const unsigned short* brow1 = &BsT[buf][wn * 32 + 16 + l16][8 * half];
    b1.h[0] = *(const u16x8*)(brow1);
    b1.h[1] = *(const u16x8*)(brow1 + 16);
    acc0 = __builtin_amdgcn_wmma_f32_16x16x32_bf16(
        false, a.v, false, b0.v, (short)0, acc0, false, false);
    acc1 = __builtin_amdgcn_wmma_f32_16x16x32_bf16(
        false, a.v, false, b1.v, (short)0, acc1, false, false);
  }
#undef GEMM_STAGE

  const int row0 = rowBase + wm * 16;
  for (int sub = 0; sub < 2; ++sub) {
    const f32x8& acc = sub ? acc1 : acc0;
    const int col = colBase + wn * 32 + sub * 16 + l16;
    const float bcol = bias[col];
    for (int r = 0; r < 8; ++r) {
      int row = row0 + r + 8 * half;
      float v = acc[r] + bcol;
      if (mode == 0) {  // GEMM1: qkv raw fp32, mlp -> gelu -> bf16 concat buf
        if (col < 3 * HID)
          h_qkv[(size_t)row * (3 * HID) + col] = v;
        else
          A2[(size_t)row * CAT + HID + (col - 3 * HID)] = f2bf(geluf(v));
      } else if (mode == 1) {  // plain fp32
        outf[(size_t)row * N + col] = v;
      } else {                 // GEMM2: residual + gate
        outf[(size_t)row * N + col] =
            xres[(size_t)row * N + col] + gate[col] * v;
      }
    }
  }
}

// ------------------- qkv post: RMS + RoPE + layout -------------------------
// grid (SEQ, NH), 128 threads. Writes head-major bf16 q/k/v + no-rope q.

__global__ __launch_bounds__(128)
void qkv_post_kernel(const float* __restrict__ h_qkv,
                     const float* __restrict__ qn_w,
                     const float* __restrict__ kn_w,
                     const float* __restrict__ fcos,
                     const float* __restrict__ fsin,
                     unsigned short* __restrict__ qr,
                     unsigned short* __restrict__ kr,
                     unsigned short* __restrict__ vb,
                     unsigned short* __restrict__ qno) {
  int s = blockIdx.x, hh = blockIdx.y, d = threadIdx.x;
  __shared__ float red[128];
  const float* base = h_qkv + (size_t)s * (3 * HID) + hh * HD;
  float q = base[d], k = base[HID + d], v = base[2 * HID + d];

  red[d] = q * q; __syncthreads();
  for (int off = 64; off > 0; off >>= 1) {
    if (d < off) red[d] += red[d + off];
    __syncthreads();
  }
  float rq = rsqrtf(red[0] * (1.0f / HD) + 1e-6f);
  __syncthreads();
  red[d] = k * k; __syncthreads();
  for (int off = 64; off > 0; off >>= 1) {
    if (d < off) red[d] += red[d + off];
    __syncthreads();
  }
  float rk = rsqrtf(red[0] * (1.0f / HD) + 1e-6f);

  float qn = q * rq * qn_w[d];
  float kn = k * rk * kn_w[d];
  float qo = qn, ko = kn;
  if (s < IMG) {
    float c = fcos[s * HD + d], si = fsin[s * HD + d];
    float qp = base[(d ^ 1)] * rq * qn_w[d ^ 1];
    float kp = base[HID + (d ^ 1)] * rk * kn_w[d ^ 1];
    float sgn = (d & 1) ? 1.f : -1.f;
    qo = qn * c + sgn * qp * si;
    ko = kn * c + sgn * kp * si;
  }
  size_t o = ((size_t)hh * SEQ + s) * HD + d;
  qr[o] = f2bf(qo); kr[o] = f2bf(ko); vb[o] = f2bf(v);
  if (s < IMG) qno[((size_t)hh * IMG + s) * HD + d] = f2bf(qn);
}

// grid (REF, NH)
__global__ __launch_bounds__(128)
void refkv_post_kernel(const float* __restrict__ rkv,
                       const float* __restrict__ kn_w,
                       const float* __restrict__ fcos,
                       const float* __restrict__ fsin,
                       unsigned short* __restrict__ rk,
                       unsigned short* __restrict__ rv) {
  int s = blockIdx.x, hh = blockIdx.y, d = threadIdx.x;
  __shared__ float red[128];
  const float* base = rkv + (size_t)s * NKV + hh * HD;
  float k = base[d], v = base[HID + d];
  red[d] = k * k; __syncthreads();
  for (int off = 64; off > 0; off >>= 1) {
    if (d < off) red[d] += red[d + off];
    __syncthreads();
  }
  float rkf = rsqrtf(red[0] * (1.0f / HD) + 1e-6f);
  float kn = k * rkf * kn_w[d];
  float c = fcos[s * HD + d], si = fsin[s * HD + d];
  float kp = base[(d ^ 1)] * rkf * kn_w[d ^ 1];
  float sgn = (d & 1) ? 1.f : -1.f;
  float ko = kn * c + sgn * kp * si;
  size_t o = ((size_t)hh * REF + s) * HD + d;
  rk[o] = f2bf(ko); rv[o] = f2bf(v);
}

// --------------------------- flash attention -------------------------------
// grid (Lq/128, NH), 256 threads = 8 waves; each wave owns a 16-row q tile.
// Block stages 32x128 K (async-to-LDS) and V^T bf16 tiles in LDS per kv step.

__global__ __launch_bounds__(256)
void flash_attn_kernel(const unsigned short* __restrict__ Q,   // [H][Lq][D]
                       const unsigned short* __restrict__ Kb,  // [H][Lk][D]
                       const unsigned short* __restrict__ Vb,  // [H][Lk][D]
                       float* __restrict__ Out,                // [Lq][H*D]
                       int Lq, int Lk, float scale) {
  __shared__ __align__(16) unsigned short Ks[32][128];
  __shared__ __align__(16) unsigned short VsT[128][32];
  __shared__ __align__(16) float Ss[8][16][32];
  __shared__ __align__(16) unsigned short Ps[8][16][32];
  __shared__ float alphaS[8][16];
  __shared__ float linvS[8][16];

  const int tid = threadIdx.x;
  const int wave = tid >> 5, lane = tid & 31;
  const int half = lane >> 4, l16 = lane & 15;
  const int head = blockIdx.y;
  const int qbase = blockIdx.x * 128 + wave * 16;

  // Q fragments for the 4 D-chunks, held in registers for the whole kv loop.
  ABFrag aq[4];
  const unsigned short* qrow = Q + ((size_t)head * Lq + qbase + l16) * HD;
  for (int c = 0; c < 4; ++c) {
    aq[c].h[0] = *(const u16x8*)(qrow + c * 32 + 8 * half);
    aq[c].h[1] = *(const u16x8*)(qrow + c * 32 + 16 + 8 * half);
  }

  f32x8 O[8];
  for (int t = 0; t < 8; ++t)
    for (int r = 0; r < 8; ++r) O[t][r] = 0.f;
  float m_run = -3.0e38f, l_run = 0.f;

  const int sIdx = tid * 16;
  const int sr = sIdx >> 7, sc = sIdx & 127;

  const int nkt = Lk / 32;
  for (int kt = 0; kt < nkt; ++kt) {
    // cooperative stage of K (row-major, async DMA) and V (transposed)
    {
      const unsigned short* krow =
          Kb + ((size_t)head * Lk + kt * 32 + sr) * HD + sc;
      copy16_to_lds(krow, &Ks[sr][sc]);
      copy16_to_lds(krow + 8, &Ks[sr][sc + 8]);
      const unsigned short* vrow =
          Vb + ((size_t)head * Lk + kt * 32 + sr) * HD + sc;
      u16x8 v0 = *(const u16x8*)(vrow);
      u16x8 v1 = *(const u16x8*)(vrow + 8);
      for (int i = 0; i < 8; ++i) {
        VsT[sc + i][sr] = v0[i];
        VsT[sc + 8 + i][sr] = v1[i];
      }
    }
    async_copy_wait();
    __syncthreads();

    // S = scale * Q K^T  (two 16x16 tiles covering 32 kv rows)
    f32x8 sacc[2];
    for (int u = 0; u < 2; ++u)
      for (int r = 0; r < 8; ++r) sacc[u][r] = 0.f;
    for (int u = 0; u < 2; ++u) {
      for (int c = 0; c < 4; ++c) {
        ABFrag bk;
        const unsigned short* krow2 = &Ks[u * 16 + l16][c * 32 + 8 * half];
        bk.h[0] = *(const u16x8*)(krow2);
        bk.h[1] = *(const u16x8*)(krow2 + 16);
        sacc[u] = __builtin_amdgcn_wmma_f32_16x16x32_bf16(
            false, aq[c].v, false, bk.v, (short)0, sacc[u], false, false);
      }
    }
    for (int r = 0; r < 8; ++r) {
      Ss[wave][r + 8 * half][l16]      = sacc[0][r] * scale;
      Ss[wave][r + 8 * half][16 + l16] = sacc[1][r] * scale;
    }
    __builtin_amdgcn_wave_barrier();  // LDS is in-order per wave

    // online softmax: lane m (< 16) owns row m of this wave's tile
    if (lane < 16) {
      int m = lane;
      float tmax = -3.0e38f;
      for (int c = 0; c < 32; ++c) tmax = fmaxf(tmax, Ss[wave][m][c]);
      float mnew = fmaxf(m_run, tmax);
      float alpha = __expf(m_run - mnew);
      float psum = 0.f;
      for (int c = 0; c < 32; ++c) {
        float p = __expf(Ss[wave][m][c] - mnew);
        psum += p;
        Ps[wave][m][c] = f2bf(p);
      }
      l_run = l_run * alpha + psum;
      m_run = mnew;
      alphaS[wave][m] = alpha;
    }
    __builtin_amdgcn_wave_barrier();

    // rescale accumulators, then O += P V
    float af[8];
    for (int r = 0; r < 8; ++r) af[r] = alphaS[wave][r + 8 * half];
    for (int t = 0; t < 8; ++t)
      for (int r = 0; r < 8; ++r) O[t][r] *= af[r];

    ABFrag pf;
    const unsigned short* prow = &Ps[wave][l16][0];
    pf.h[0] = *(const u16x8*)(prow + 8 * half);
    pf.h[1] = *(const u16x8*)(prow + 16 + 8 * half);
    for (int t = 0; t < 8; ++t) {
      ABFrag vf;
      const unsigned short* vrow2 = &VsT[t * 16 + l16][8 * half];
      vf.h[0] = *(const u16x8*)(vrow2);
      vf.h[1] = *(const u16x8*)(vrow2 + 16);
      O[t] = __builtin_amdgcn_wmma_f32_16x16x32_bf16(
          false, pf.v, false, vf.v, (short)0, O[t], false, false);
    }
    __syncthreads();
  }

  if (lane < 16) linvS[wave][lane] = 1.0f / l_run;
  __builtin_amdgcn_wave_barrier();
  float li[8];
  for (int r = 0; r < 8; ++r) li[r] = linvS[wave][r + 8 * half];
  for (int t = 0; t < 8; ++t) {
    for (int r = 0; r < 8; ++r) {
      int row = qbase + r + 8 * half;
      int col = head * HD + t * 16 + l16;
      Out[(size_t)row * (NH * HD) + col] = O[t][r] * li[r];
    }
  }
}

// attn(+ip for img rows) -> bf16 into concat buffer columns [0, HID)
__global__ void build_a2_kernel(const float* __restrict__ attn,
                                const float* __restrict__ ip,
                                unsigned short* __restrict__ A2) {
  int idx = blockIdx.x * 256 + threadIdx.x;
  int s = idx / HID, c = idx - s * HID;
  float v = attn[idx];
  if (s < IMG) v += ip[(size_t)s * HID + c];
  A2[(size_t)s * CAT + c] = f2bf(v);
}

// ------------------------------- launcher ----------------------------------

extern "C" void kernel_launch(void* const* d_in, const int* in_sizes, int n_in,
                              void* d_out, int out_size, void* d_ws,
                              size_t ws_size, hipStream_t stream) {
  const float* x       = (const float*)d_in[0];
  const float* vec     = (const float*)d_in[1];
  const float* W1      = (const float*)d_in[2];
  const float* b1      = (const float*)d_in[3];
  const float* W2      = (const float*)d_in[4];
  const float* b2      = (const float*)d_in[5];
  const float* mod_W   = (const float*)d_in[6];
  const float* mod_b   = (const float*)d_in[7];
  const float* qn_w    = (const float*)d_in[8];
  const float* kn_w    = (const float*)d_in[9];
  const float* fcos    = (const float*)d_in[10];
  const float* fsin    = (const float*)d_in[11];
  const float* ref_lat = (const float*)d_in[12];
  const float* ref_Wkv = (const float*)d_in[13];
  const float* ref_bkv = (const float*)d_in[14];
  const float* ref_knw = (const float*)d_in[15];
  const float* ref_cos = (const float*)d_in[16];
  const float* ref_sin = (const float*)d_in[17];
  float* out = (float*)d_out;

  char* w = (char*)d_ws;
  size_t off = 0;
  auto alloc = [&](size_t bytes) -> void* {
    off = (off + 255) & ~(size_t)255;
    void* p = w + off;
    off += bytes;
    return p;
  };

  float* sv    = (float*)alloc((size_t)HID * 4);
  float* mod   = (float*)alloc((size_t)3 * HID * 4);
  unsigned short* xmod = (unsigned short*)alloc((size_t)SEQ * HID * 2);
  float* h_qkv = (float*)alloc((size_t)SEQ * 3 * HID * 4);
  unsigned short* A2 = (unsigned short*)alloc((size_t)SEQ * CAT * 2);
  unsigned short* qr = (unsigned short*)alloc((size_t)NH * SEQ * HD * 2);
  unsigned short* kr = (unsigned short*)alloc((size_t)NH * SEQ * HD * 2);
  unsigned short* vb = (unsigned short*)alloc((size_t)NH * SEQ * HD * 2);
  unsigned short* qno = (unsigned short*)alloc((size_t)NH * IMG * HD * 2);
  float* rkv = (float*)alloc((size_t)REF * NKV * 4);
  unsigned short* rAbf = (unsigned short*)alloc((size_t)REF * HID * 2);
  unsigned short* rk = (unsigned short*)alloc((size_t)NH * REF * HD * 2);
  unsigned short* rv = (unsigned short*)alloc((size_t)NH * REF * HD * 2);
  float* attn = (float*)alloc((size_t)SEQ * HID * 4);
  float* ip   = (float*)alloc((size_t)IMG * HID * 4);

  const float scale = 0.08838834764831845f;  // 1/sqrt(128)

  // 1) modulation
  silu_kernel<<<HID / 256, 256, 0, stream>>>(vec, sv);
  mod_gemv_kernel<<<(3 * HID) / 256, 256, 0, stream>>>(sv, mod_W, mod_b, mod);
  // 2) layernorm + modulate -> bf16
  ln_mod_kernel<<<SEQ, 256, 0, stream>>>(x, mod, xmod);
  // 3) h = x_mod @ W1 + b1 (qkv fp32 | gelu(mlp) bf16)
  gemm_bf16_wmma_kernel<<<dim3(SEQ / 64, N1 / 64), 256, 0, stream>>>(
      xmod, W1, b1, SEQ, N1, HID, 0, nullptr, nullptr, nullptr, h_qkv, A2);
  // 4) q/k RMS + RoPE + head-major bf16 layout
  qkv_post_kernel<<<dim3(SEQ, NH), 128, 0, stream>>>(
      h_qkv, qn_w, kn_w, fcos, fsin, qr, kr, vb, qno);
  // 5) self attention
  flash_attn_kernel<<<dim3(SEQ / 128, NH), 256, 0, stream>>>(
      qr, kr, vb, attn, SEQ, SEQ, scale);
  // 6) ref kv projection + post + cross attention
  f32_to_bf16_kernel<<<(REF * HID) / 256, 256, 0, stream>>>(ref_lat, rAbf);
  gemm_bf16_wmma_kernel<<<dim3(REF / 64, NKV / 64), 256, 0, stream>>>(
      rAbf, ref_Wkv, ref_bkv, REF, NKV, HID, 1, rkv, nullptr, nullptr,
      nullptr, nullptr);
  refkv_post_kernel<<<dim3(REF, NH), 128, 0, stream>>>(
      rkv, ref_knw, ref_cos, ref_sin, rk, rv);
  flash_attn_kernel<<<dim3(IMG / 128, NH), 256, 0, stream>>>(
      qno, rk, rv, ip, IMG, REF, scale);
  // 7) concat attn(+ip) into A2 cols [0,HID)
  build_a2_kernel<<<(SEQ * HID) / 256, 256, 0, stream>>>(attn, ip, A2);
  // 8) out = x + gate * (A2 @ W2 + b2)
  gemm_bf16_wmma_kernel<<<dim3(SEQ / 64, HID / 64), 256, 0, stream>>>(
      A2, W2, b2, SEQ, HID, CAT, 2, out, x, mod + 2 * HID, nullptr, nullptr);
}
